// PairformerBlock_86019605004409
// MI455X (gfx1250) — compile-verified, matrix operand order checked by reading, not run on hardware
//
#include <hip/hip_runtime.h>

typedef _Float16 h16;
typedef __attribute__((ext_vector_type(16))) _Float16 v16h;
typedef __attribute__((ext_vector_type(8)))  _Float16 v8h;
typedef __attribute__((ext_vector_type(8)))  float    v8f;

#define DEVFN static __device__ __forceinline__

// CDNA5 16-bit A-operand fragment: per lane (m = lane&15, hf = lane>>4) the
// 16 elements are two contiguous 8-half runs at rp+8*hf and rp+8*hf+16.
DEVFN v16h load_frag(const h16* rp, int hf) {
  const h16* p = rp + 8 * hf;
  v8h lo = *(const v8h*)p;
  v8h hi = *(const v8h*)(p + 16);
  v16h a;
#pragma unroll
  for (int e = 0; e < 8; ++e) a[e] = lo[e];
#pragma unroll
  for (int e = 0; e < 8; ++e) a[8 + e] = hi[e];
  return a;
}

DEVFN v16h load_frag_lds_f32(const float* rp, int hf) {
  const float* p = rp + 8 * hf;
  v16h a;
#pragma unroll
  for (int e = 0; e < 8; ++e) a[e] = (h16)p[e];
#pragma unroll
  for (int e = 0; e < 8; ++e) a[8 + e] = (h16)p[16 + e];
  return a;
}

DEVFN v8f wmma_f16(v16h a, v16h b, v8f c) {
  return __builtin_amdgcn_wmma_f32_16x16x32_f16(false, a, false, b, (short)0, c,
                                                false, false);
}

DEVFN v8f v8f_zero() {
  v8f z;
#pragma unroll
  for (int i = 0; i < 8; ++i) z[i] = 0.f;
  return z;
}

DEVFN float sigm(float x) { return 1.f / (1.f + __expf(-x)); }

// ---- generic GEMM: C[MxN] = A[MxK] @ W[KxN], W pre-transposed [N][K] ----
// One wave computes a 16x64 strip (4 tiles); A fragment reused across 4 WMMAs.
__global__ void gemm16(const h16* __restrict__ A, const h16* __restrict__ WT,
                       float* Cf, h16* Ch, int M, int N, int K, int lda,
                       int ldc) {
  int ns = (N + 63) >> 6;
  int wid = threadIdx.x >> 5;
  int t = blockIdx.x * 4 + wid;
  if (t >= ns * (M >> 4)) return;  // wave-uniform
  int s0 = (t % ns) * 64;
  int m0 = (t / ns) * 16;
  int lane = threadIdx.x & 31;
  int ml = lane & 15, hf = lane >> 4;
  const h16* arow = A + (long)(m0 + ml) * lda;
  const h16* wrow[4];
#pragma unroll
  for (int u = 0; u < 4; ++u) {
    int nn = s0 + u * 16 + ml;
    if (nn >= N) nn = N - 1;  // clamp; unstored lanes compute dup columns
    wrow[u] = WT + (long)nn * K;
  }
  v8f acc[4];
#pragma unroll
  for (int u = 0; u < 4; ++u) acc[u] = v8f_zero();
  for (int k0 = 0; k0 < K; k0 += 32) {
    v16h a = load_frag(arow + k0, hf);
#pragma unroll
    for (int u = 0; u < 4; ++u) {
      v16h b = load_frag(wrow[u] + k0, hf);
      acc[u] = wmma_f16(a, b, acc[u]);
    }
  }
#pragma unroll
  for (int u = 0; u < 4; ++u) {
    int n = s0 + u * 16 + ml;
    if (n < N) {
#pragma unroll
      for (int r = 0; r < 8; ++r) {
        long m = m0 + r + 8 * hf;
        if (Ch) Ch[m * ldc + n] = (h16)acc[u][r];
        else    Cf[m * ldc + n] = acc[u][r];
      }
    }
  }
}

// Fused transition hidden: H = silu(A@W1) * (A@W2), f16 out; 16x64 strip,
// A fragment reused across 8 WMMAs.
__global__ void gemm16_dual(const h16* __restrict__ A,
                            const h16* __restrict__ W1T,
                            const h16* __restrict__ W2T, h16* __restrict__ Ch,
                            int M, int N, int K, int lda, int ldc) {
  int ns = N >> 6;
  int wid = threadIdx.x >> 5;
  int t = blockIdx.x * 4 + wid;
  if (t >= ns * (M >> 4)) return;
  int s0 = (t % ns) * 64;
  int m0 = (t / ns) * 16;
  int lane = threadIdx.x & 31;
  int ml = lane & 15, hf = lane >> 4;
  const h16* arow = A + (long)(m0 + ml) * lda;
  const h16 *w1r[4], *w2r[4];
#pragma unroll
  for (int u = 0; u < 4; ++u) {
    int nn = s0 + u * 16 + ml;
    w1r[u] = W1T + (long)nn * K;
    w2r[u] = W2T + (long)nn * K;
  }
  v8f a1[4], a2[4];
#pragma unroll
  for (int u = 0; u < 4; ++u) { a1[u] = v8f_zero(); a2[u] = v8f_zero(); }
  for (int k0 = 0; k0 < K; k0 += 32) {
    v16h a = load_frag(arow + k0, hf);
#pragma unroll
    for (int u = 0; u < 4; ++u) {
      v16h b1 = load_frag(w1r[u] + k0, hf);
      a1[u] = wmma_f16(a, b1, a1[u]);
      v16h b2 = load_frag(w2r[u] + k0, hf);
      a2[u] = wmma_f16(a, b2, a2[u]);
    }
  }
#pragma unroll
  for (int u = 0; u < 4; ++u) {
#pragma unroll
    for (int r = 0; r < 8; ++r) {
      long m = m0 + r + 8 * hf;
      float x = a1[u][r], y = a2[u][r];
      Ch[m * ldc + s0 + u * 16 + ml] = (h16)((x * sigm(x)) * y);
    }
  }
}

// O[(i*256+j)*128 + c] = sum_k At[c][i][k] * Bt[c][j][k]
// Wave computes 16(i) x 64(j); A fragment reused across 4 WMMAs.
__global__ void tri_einsum_k(const h16* __restrict__ At,
                             const h16* __restrict__ Bt,
                             float* __restrict__ O) {
  int c = blockIdx.z;
  int wid = threadIdx.x >> 5;
  int i0 = blockIdx.y * 16;
  int j0 = wid * 64;
  int lane = threadIdx.x & 31;
  int ml = lane & 15, hf = lane >> 4;
  const h16* arow = At + c * 65536 + (i0 + ml) * 256;
  const h16* brow[4];
#pragma unroll
  for (int u = 0; u < 4; ++u) brow[u] = Bt + c * 65536 + (j0 + u * 16 + ml) * 256;
  v8f acc[4];
#pragma unroll
  for (int u = 0; u < 4; ++u) acc[u] = v8f_zero();
  for (int k0 = 0; k0 < 256; k0 += 32) {
    v16h a = load_frag(arow + k0, hf);
#pragma unroll
    for (int u = 0; u < 4; ++u) {
      v16h b = load_frag(brow[u] + k0, hf);
      acc[u] = wmma_f16(a, b, acc[u]);
    }
  }
#pragma unroll
  for (int u = 0; u < 4; ++u) {
#pragma unroll
    for (int r = 0; r < 8; ++r) {
      int i = i0 + r + 8 * hf, j = j0 + u * 16 + ml;
      O[((long)i * 256 + j) * 128 + c] = acc[u][r];
    }
  }
}

// ---------------- fused attention (4 query tiles per block) ----------------
__global__ void attn_fused(const h16* __restrict__ Q, const h16* __restrict__ Km,
                           const h16* __restrict__ Vt,
                           const float* __restrict__ Bias,
                           const unsigned char* __restrict__ Mask,
                           h16* __restrict__ O, int H, long qOuter, int qRow,
                           int msOuter, int msK, float scale) {
  __shared__ float lgs[4 * 16 * 256];
  int wid = threadIdx.x >> 5;
  float* lg = lgs + wid * (16 * 256);
  int b = blockIdx.x;
  int outer = b / H, h = b % H;
  int j0 = (blockIdx.y * 4 + wid) * 16;
  int lane = threadIdx.x & 31;
  int ml = lane & 15, hf = lane >> 4;

  const h16* qb = Q + (long)outer * qOuter + h * 32;
  const h16* kb = Km + (long)outer * qOuter + h * 32;
  const h16* vb = Vt + ((long)(outer * H + h) * 32) * 256;
  const float* bb = Bias + h;

  v16h aQ = load_frag(qb + (j0 + ml) * qRow, hf);
  const h16* krow = kb + ml * qRow;
  for (int k0 = 0; k0 < 256; k0 += 16) {
    v16h bK = load_frag(krow, hf);
    krow += 16 * qRow;
    v8f acc = v8f_zero();
    acc = wmma_f16(aQ, bK, acc);
#pragma unroll
    for (int r = 0; r < 8; ++r) {
      int jl = r + 8 * hf;
      int kg = k0 + ml;
      float bias = bb[((j0 + jl) * 256 + kg) * H];
      float mv = (Mask[outer * msOuter + kg * msK] > 0) ? 0.f : -1e9f;
      lg[jl * 256 + kg] = acc[r] * scale + bias + mv;
    }
  }
  __syncthreads();

  int row = lane >> 1, side = lane & 1;
  float mx = -1e30f;
  for (int t = 0; t < 128; ++t) mx = fmaxf(mx, lg[row * 256 + side * 128 + t]);
  mx = fmaxf(mx, __shfl_xor(mx, 1));
  float sum = 0.f;
  for (int t = 0; t < 128; ++t) {
    int idx = row * 256 + side * 128 + t;
    float e = __expf(lg[idx] - mx);
    lg[idx] = e;
    sum += e;
  }
  sum += __shfl_xor(sum, 1);
  float inv = 1.f / sum;
  for (int t = 0; t < 128; ++t) lg[row * 256 + side * 128 + t] *= inv;
  __syncthreads();

  // O tile = P(16x256) @ V(256x32); P fragment reused for both c halves.
  v8f acc0 = v8f_zero(), acc1 = v8f_zero();
  const float* prow = lg + ml * 256;
  const h16* vrow0 = vb + ml * 256;
  const h16* vrow1 = vb + (16 + ml) * 256;
  for (int k0 = 0; k0 < 256; k0 += 32) {
    v16h a = load_frag_lds_f32(prow + k0, hf);
    v16h b0 = load_frag(vrow0 + k0, hf);
    acc0 = wmma_f16(a, b0, acc0);
    v16h b1 = load_frag(vrow1 + k0, hf);
    acc1 = wmma_f16(a, b1, acc1);
  }
  h16* ob = O + (long)outer * 256 * (H * 32) + h * 32;
#pragma unroll
  for (int r = 0; r < 8; ++r) {
    int jl = r + 8 * hf;
    ob[(j0 + jl) * (H * 32) + ml] = (h16)acc0[r];
    ob[(j0 + jl) * (H * 32) + 16 + ml] = (h16)acc1[r];
  }
}

// ---------------- LayerNorm: f32 rows -> f16 rows ----------------
__global__ void ln_f16_k(const float* __restrict__ X, const float* __restrict__ g,
                         const float* __restrict__ b, h16* __restrict__ Y, int C,
                         int trans) {
  int wid = threadIdx.x >> 5, lane = threadIdx.x & 31;
  int row = blockIdx.x * 8 + wid;
  const float* x = X + (long)row * C;
  float s = 0.f, s2 = 0.f;
  for (int c = lane; c < C; c += 32) {
    float v = x[c];
    s += v;
    s2 += v * v;
  }
#pragma unroll
  for (int off = 16; off; off >>= 1) {
    s += __shfl_xor(s, off);
    s2 += __shfl_xor(s2, off);
  }
  float mu = s / C;
  float var = s2 / C - mu * mu;
  float rstd = rsqrtf(var + 1e-5f);
  int orow = trans ? (((row & 255) << 8) | (row >> 8)) : row;
  h16* y = Y + (long)orow * C;
  for (int c = lane; c < C; c += 32)
    y[c] = (h16)(((x[c] - mu) * rstd) * g[c] + b[c]);
}

// ---------------- elementwise ----------------
__global__ void cvt_t_k(const float* __restrict__ x, h16* __restrict__ y, int K,
                        int N) {
  int n = K * N;
  for (int i = blockIdx.x * blockDim.x + threadIdx.x; i < n;
       i += gridDim.x * blockDim.x) {
    int k = i / N, c = i - k * N;
    y[c * K + k] = (h16)x[i];
  }
}

__global__ void combine_tri_k(h16* __restrict__ dst, const float* __restrict__ P1,
                              const float* __restrict__ P2,
                              const unsigned char* __restrict__ pm, int trans,
                              int n) {
  for (int idx = blockIdx.x * blockDim.x + threadIdx.x; idx < n;
       idx += gridDim.x * blockDim.x) {
    int c = idx >> 16;
    int rest = idx & 65535;
    int i = rest >> 8, k = rest & 255;
    int sp = trans ? ((k << 8) | i) : rest;
    float v = sigm(P1[sp * 128 + c]) * P2[sp * 128 + c];
    v *= pm[sp] ? 1.f : 0.f;
    dst[idx] = (h16)v;
  }
}

__global__ void sig_mul_add_k(float* __restrict__ z, const float* __restrict__ G,
                              const float* __restrict__ P, int n) {
  for (int i = blockIdx.x * blockDim.x + threadIdx.x; i < n;
       i += gridDim.x * blockDim.x)
    z[i] += sigm(G[i]) * P[i];
}

__global__ void gate_mul_h_k(h16* __restrict__ dst, const float* __restrict__ G,
                             const h16* __restrict__ Oh, int n) {
  for (int i = blockIdx.x * blockDim.x + threadIdx.x; i < n;
       i += gridDim.x * blockDim.x)
    dst[i] = (h16)(sigm(G[i]) * (float)Oh[i]);
}

__global__ void res_add_k(float* __restrict__ d, const float* __restrict__ s,
                          int n) {
  for (int i = blockIdx.x * blockDim.x + threadIdx.x; i < n;
       i += gridDim.x * blockDim.x)
    d[i] += s[i];
}

__global__ void res_add_T_k(float* __restrict__ z, const float* __restrict__ F,
                            int n) {
  for (int idx = blockIdx.x * blockDim.x + threadIdx.x; idx < n;
       idx += gridDim.x * blockDim.x) {
    int c = idx & 127;
    int sp = idx >> 7;
    int i = sp >> 8, j = sp & 255;
    z[idx] += F[((((j << 8) | i)) << 7) + c];
  }
}

__global__ void mask_rows_k(float* __restrict__ x,
                            const unsigned char* __restrict__ m, int shift,
                            int n) {
  for (int i = blockIdx.x * blockDim.x + threadIdx.x; i < n;
       i += gridDim.x * blockDim.x)
    x[i] *= m[i >> shift] ? 1.f : 0.f;
}

__global__ void transpose_v_k(h16* __restrict__ vt, const h16* __restrict__ v,
                              int H, int n) {
  for (int idx = blockIdx.x * blockDim.x + threadIdx.x; idx < n;
       idx += gridDim.x * blockDim.x) {
    int k = idx & 255;
    int c = (idx >> 8) & 31;
    int h = (idx >> 13) % H;
    int outer = idx / (8192 * H);
    vt[idx] = v[((outer * 256 + k) * H + h) * 32 + c];
  }
}

// ---------------- host orchestration ----------------
extern "C" void kernel_launch(void* const* d_in, const int* in_sizes, int n_in,
                              void* d_out, int out_size, void* d_ws,
                              size_t ws_size, hipStream_t stream) {
  (void)n_in; (void)out_size; (void)ws_size; (void)in_sizes;
  const int CZ = 128, CS = 256;
  const long NZ = 65536;
  const long ZE = NZ * CZ;

  const float* s_in = (const float*)d_in[0];
  const float* z_in = (const float*)d_in[1];
  const unsigned char* rm = (const unsigned char*)d_in[2];
  const unsigned char* pm = (const unsigned char*)d_in[3];
  auto P = [&](int i) { return (const float*)d_in[i]; };

  char* base = (char*)d_ws;
  size_t off = 0;
  auto alloc = [&](size_t bytes) -> void* {
    void* p = base + off;
    off = (off + bytes + 255) & ~(size_t)255;
    return p;
  };
  float* zc = (float*)alloc(ZE * 4);
  float* sc = (float*)alloc(NZ * 4);
  h16* XH = (h16*)alloc(ZE * 2);
  h16* H1 = (h16*)alloc(ZE * 2);
  h16* H2 = (h16*)alloc(ZE * 2);
  h16* H3 = (h16*)alloc(ZE * 2);
  h16* H4 = (h16*)alloc(ZE * 2);
  float* F0 = (float*)alloc(ZE * 4);
  float* F1 = (float*)alloc(ZE * 4);
  float* F2 = (float*)alloc(ZE * 4);
  h16* HT = (h16*)F0;  // 65536x512 f16 transition hidden overlays F0+F1
  h16* SH = (h16*)alloc(NZ * 2);
  float* BIAS = (float*)alloc(NZ * 8 * 4);
  char* wslab = (char*)alloc((size_t)4 << 20);

  size_t woff = 0;
  // convert f32 [K,N] weight -> f16 transposed [N,K]
  auto cw = [&](int idx, int K, int N) -> h16* {
    int n = K * N;
    h16* p = (h16*)(wslab + woff);
    woff = (woff + (size_t)n * 2 + 255) & ~(size_t)255;
    int blocks = (n + 255) / 256;
    if (blocks > 2048) blocks = 2048;
    cvt_t_k<<<blocks, 256, 0, stream>>>(P(idx), p, K, N);
    return p;
  };

  h16 *tmo_ag = cw(6, 128, 128), *tmo_ap = cw(7, 128, 128),
      *tmo_bg = cw(8, 128, 128), *tmo_bp = cw(9, 128, 128),
      *tmo_g = cw(10, 128, 128), *tmo_out = cw(13, 128, 128);
  h16 *tmi_ag = cw(16, 128, 128), *tmi_ap = cw(17, 128, 128),
      *tmi_bg = cw(18, 128, 128), *tmi_bp = cw(19, 128, 128),
      *tmi_g = cw(20, 128, 128), *tmi_out = cw(23, 128, 128);
  h16 *tas_wq = cw(26, 128, 128), *tas_wk = cw(27, 128, 128),
      *tas_wv = cw(28, 128, 128), *tas_wb = cw(29, 128, 4),
      *tas_wg = cw(30, 128, 128), *tas_wo = cw(31, 128, 128);
  h16 *tae_wq = cw(34, 128, 128), *tae_wk = cw(35, 128, 128),
      *tae_wv = cw(36, 128, 128), *tae_wb = cw(37, 128, 4),
      *tae_wg = cw(38, 128, 128), *tae_wo = cw(39, 128, 128);
  h16 *trz_w1 = cw(42, 128, 512), *trz_w2 = cw(43, 128, 512),
      *trz_w3 = cw(44, 512, 128);
  h16 *apb_wq = cw(49, 256, 256), *apb_wk = cw(50, 256, 256),
      *apb_wv = cw(51, 256, 256), *apb_wb = cw(52, 128, 8),
      *apb_wg = cw(53, 256, 256), *apb_wo = cw(54, 256, 256);
  h16 *trs_w1 = cw(57, 256, 1024), *trs_w2 = cw(58, 256, 1024),
      *trs_w3 = cw(59, 1024, 256);

  hipMemcpyAsync(zc, z_in, ZE * 4, hipMemcpyDeviceToDevice, stream);
  hipMemcpyAsync(sc, s_in, NZ * 4, hipMemcpyDeviceToDevice, stream);

  auto LN = [&](const float* X, long rows, int C, const float* g,
                const float* b, h16* Y, int trans) {
    ln_f16_k<<<(int)(rows / 8), 256, 0, stream>>>(X, g, b, Y, C, trans);
  };
  auto GEMM = [&](const h16* A, const h16* WT, float* Cf, h16* Ch, int M, int N,
                  int K, int lda, int ldc) {
    int ns = (N + 63) >> 6;
    int tiles = ns * (M >> 4);
    gemm16<<<(tiles + 3) / 4, 128, 0, stream>>>(A, WT, Cf, Ch, M, N, K, lda,
                                                ldc);
  };
  auto EW = [&](long n) {
    int b = (int)((n + 255) / 256);
    return b > 8192 ? 8192 : b;
  };
  const float SCALE = 0.17677669529663687f;  // 1/sqrt(32)

  auto TRIMUL = [&](int pb, h16* ag, h16* ap, h16* bg_, h16* bp_, h16* gw,
                    h16* ow, int incoming) {
    LN(zc, NZ, CZ, P(pb), P(pb + 1), XH, 0);
    GEMM(XH, ag, F0, nullptr, 65536, 128, 128, 128, 128);
    GEMM(XH, ap, F1, nullptr, 65536, 128, 128, 128, 128);
    combine_tri_k<<<EW(ZE), 256, 0, stream>>>(H1, F0, F1, pm, incoming, (int)ZE);
    GEMM(XH, bg_, F0, nullptr, 65536, 128, 128, 128, 128);
    GEMM(XH, bp_, F1, nullptr, 65536, 128, 128, 128, 128);
    combine_tri_k<<<EW(ZE), 256, 0, stream>>>(H2, F0, F1, pm, incoming, (int)ZE);
    tri_einsum_k<<<dim3(1, 16, 128), 128, 0, stream>>>(H1, H2, F0);
    LN(F0, NZ, CZ, P(pb + 7), P(pb + 8), H3, 0);
    GEMM(XH, gw, F1, nullptr, 65536, 128, 128, 128, 128);
    GEMM(H3, ow, F2, nullptr, 65536, 128, 128, 128, 128);
    sig_mul_add_k<<<EW(ZE), 256, 0, stream>>>(zc, F1, F2, (int)ZE);
  };
  TRIMUL(4, tmo_ag, tmo_ap, tmo_bg, tmo_bp, tmo_g, tmo_out, 0);
  TRIMUL(14, tmi_ag, tmi_ap, tmi_bg, tmi_bp, tmi_g, tmi_out, 1);

  auto TRIATTN = [&](int pb, h16* wq, h16* wk, h16* wv, h16* wb, h16* wg,
                     h16* wo, int ending) {
    LN(zc, NZ, CZ, P(pb), P(pb + 1), XH, ending);
    GEMM(XH, wq, nullptr, H1, 65536, 128, 128, 128, 128);
    GEMM(XH, wk, nullptr, H2, 65536, 128, 128, 128, 128);
    GEMM(XH, wv, nullptr, H3, 65536, 128, 128, 128, 128);
    transpose_v_k<<<EW(ZE), 256, 0, stream>>>(H4, H3, 4, (int)ZE);
    GEMM(XH, wb, BIAS, nullptr, 65536, 4, 128, 128, 4);
    attn_fused<<<dim3(1024, 4), 128, 0, stream>>>(
        H1, H2, H4, BIAS, pm, H3, 4, (long)256 * 128, 128, ending ? 1 : 256,
        ending ? 256 : 1, SCALE);
    GEMM(XH, wg, F1, nullptr, 65536, 128, 128, 128, 128);
    gate_mul_h_k<<<EW(ZE), 256, 0, stream>>>(H1, F1, H3, (int)ZE);
    GEMM(H1, wo, F2, nullptr, 65536, 128, 128, 128, 128);
    if (ending)
      res_add_T_k<<<EW(ZE), 256, 0, stream>>>(zc, F2, (int)ZE);
    else
      res_add_k<<<EW(ZE), 256, 0, stream>>>(zc, F2, (int)ZE);
  };
  TRIATTN(24, tas_wq, tas_wk, tas_wv, tas_wb, tas_wg, tas_wo, 0);
  TRIATTN(32, tae_wq, tae_wk, tae_wv, tae_wb, tae_wg, tae_wo, 1);

  // transition on z
  LN(zc, NZ, CZ, P(40), P(41), XH, 0);
  gemm16_dual<<<(8 * 4096) / 4, 128, 0, stream>>>(XH, trz_w1, trz_w2, HT, 65536,
                                                  512, 128, 128, 512);
  GEMM(HT, trz_w3, F2, nullptr, 65536, 128, 512, 512, 128);
  res_add_k<<<EW(ZE), 256, 0, stream>>>(zc, F2, (int)ZE);
  mask_rows_k<<<EW(ZE), 256, 0, stream>>>(zc, pm, 7, (int)ZE);

  // attention with pair bias on s
  LN(sc, 256, CS, P(45), P(46), SH, 0);
  GEMM(SH, apb_wq, nullptr, H1, 256, 256, 256, 256, 256);
  GEMM(SH, apb_wk, nullptr, H2, 256, 256, 256, 256, 256);
  GEMM(SH, apb_wv, nullptr, H3, 256, 256, 256, 256, 256);
  transpose_v_k<<<EW(NZ), 256, 0, stream>>>(H4, H3, 8, (int)NZ);
  LN(zc, NZ, CZ, P(47), P(48), XH, 0);
  GEMM(XH, apb_wb, BIAS, nullptr, 65536, 8, 128, 128, 8);
  attn_fused<<<dim3(8, 4), 128, 0, stream>>>(H1, H2, H4, BIAS, rm, H3, 8, 0L,
                                             256, 0, 1, SCALE);
  GEMM(SH, apb_wg, F1, nullptr, 256, 256, 256, 256, 256);
  gate_mul_h_k<<<EW(NZ), 256, 0, stream>>>(H1, F1, H3, (int)NZ);
  GEMM(H1, apb_wo, F2, nullptr, 256, 256, 256, 256, 256);
  res_add_k<<<EW(NZ), 256, 0, stream>>>(sc, F2, (int)NZ);

  // transition on s
  LN(sc, 256, CS, P(55), P(56), SH, 0);
  gemm16_dual<<<(16 * 16) / 4, 128, 0, stream>>>(SH, trs_w1, trs_w2, H1, 256,
                                                 1024, 256, 256, 1024);
  GEMM(H1, trs_w3, F2, nullptr, 256, 256, 1024, 1024, 256);
  res_add_k<<<EW(NZ), 256, 0, stream>>>(sc, F2, (int)NZ);
  mask_rows_k<<<EW(NZ), 256, 0, stream>>>(sc, rm, 8, (int)NZ);

  float* out = (float*)d_out;
  hipMemcpyAsync(out, sc, NZ * 4, hipMemcpyDeviceToDevice, stream);
  hipMemcpyAsync(out + NZ, zc, ZE * 4, hipMemcpyDeviceToDevice, stream);
}
